// SwinTransformerBlock_1803886264577
// MI455X (gfx1250) — compile-verified, matrix operand order checked by reading, not run on hardware
//
#include <hip/hip_runtime.h>

#define DEV __device__ __forceinline__

constexpr int DIMc   = 384;
constexpr int HEADSc = 12;
constexpr int HDIMc  = 32;
constexpr int Lc     = 49;     // tokens per 7x7 window
constexpr int LPc    = 64;     // padded
constexpr int NWc    = 64;     // windows per image (8x8)
constexpr int NBc    = 32;     // batch
constexpr int WINSc  = NBc * NWc;   // 2048
constexpr int TOKc   = WINSc * Lc;  // 100352
constexpr int QKVNc  = 3 * DIMc;    // 1152
constexpr int HIDc   = 4 * DIMc;    // 1536

// gfx1250 async global->LDS path (ASYNCcnt), guarded so absence degrades cleanly.
#if defined(__HIP_DEVICE_COMPILE__) && defined(__gfx1250__) && \
    __has_builtin(__builtin_amdgcn_global_load_async_to_lds_b128) && \
    __has_builtin(__builtin_amdgcn_s_wait_asynccnt)
#define USE_ASYNC_LDS 1
#else
#define USE_ASYNC_LDS 0
#endif

typedef __attribute__((ext_vector_type(16))) _Float16 v16h;
typedef __attribute__((ext_vector_type(8)))  float    v8f;
typedef __attribute__((ext_vector_type(4)))  int      i32x4_t;

#if USE_ASYNC_LDS
typedef __attribute__((address_space(1))) i32x4_t* gint4p;
typedef __attribute__((address_space(3))) i32x4_t* lint4p;
#endif

union HFrag { v16h v; unsigned short u[16]; };

DEV unsigned short h2u(_Float16 h) { union { _Float16 h; unsigned short u; } c; c.h = h; return c.u; }
DEV _Float16 u2h(unsigned short u) { union { _Float16 h; unsigned short u; } c; c.u = u; return c.h; }

DEV uint2 pack4h(float4 f) {
  union { _Float16 h[4]; uint2 u; } c;
  c.h[0] = (_Float16)f.x; c.h[1] = (_Float16)f.y;
  c.h[2] = (_Float16)f.z; c.h[3] = (_Float16)f.w;
  return c.u;
}

DEV v8f v8f_zero() {
  v8f z;
#pragma unroll
  for (int i = 0; i < 8; ++i) z[i] = 0.f;
  return z;
}

DEV v8f wmma16(v16h a, v16h b, v8f c) {
  return __builtin_amdgcn_wmma_f32_16x16x32_f16(false, a, false, b, (short)0, c, false, false);
}

// Fragment loader from LDS, matrix row-major [16][stride] (16-bit elements).
// Works for A tiles (rows = M) and for transposed-staged B tiles (rows = N).
// Per ISA 7.12.2: lanes 0-15 pick the row, lane>>4 = h takes k 8h..8h+7 and 16+8h..16+8h+7; the 16
// elements land so consecutive u[] entries are contiguous in LDS -> ds_load_b128.
DEV v16h frag_A(const unsigned short* p, int stride) {
  int lane = threadIdx.x & 31;
  int m = lane & 15, hf = lane >> 4;
  HFrag f;
  const unsigned short* row = p + m * stride;
#pragma unroll
  for (int vg = 0; vg < 8; ++vg) {
    int k = ((vg & 4) << 2) + (hf << 3) + ((vg & 3) << 1);
    f.u[2 * vg]     = row[k];
    f.u[2 * vg + 1] = row[k + 1];
  }
  return f.v;
}

// window-token t -> spatial row index (after cyclic shift by -3,-3)
DEV int tok_srow(int t) {
  int w = t / Lc, l = t % Lc;
  int b = w / NWc, wi = w % NWc;
  int wy = wi >> 3, wx = wi & 7;
  int ly = l / 7, lx = l % 7;
  int sy = (wy * 7 + ly + 3) % 56;
  int sx = (wx * 7 + lx + 3) % 56;
  return (b * 56 + sy) * 56 + sx;
}

// ------------------------------------------------ convert + transpose: dst[n*K+k] = (f16)src[k*N+n]
__global__ void k_cvtT(const float* __restrict__ src, _Float16* __restrict__ dst, int K, int N) {
  int i = blockIdx.x * blockDim.x + threadIdx.x;
  if (i >= K * N) return;
  int n = i / K, k = i % K;
  dst[i] = (_Float16)src[k * N + n];
}

// ------------------------------------------------ meta-MLP bias table (heads,L,L)
__global__ void k_meta_bias(const float* __restrict__ w1, const float* __restrict__ b1,
                            const float* __restrict__ w2, const float* __restrict__ b2,
                            float* __restrict__ bias) {
  int p = blockIdx.x * blockDim.x + threadIdx.x;
  if (p >= Lc * Lc) return;
  int i = p / Lc, j = p % Lc;
  float dh = (float)(i / 7 - j / 7);
  float dw = (float)(i % 7 - j % 7);
  float r0 = (dh > 0.f ? 1.f : (dh < 0.f ? -1.f : 0.f)) * log1pf(fabsf(dh));
  float r1 = (dw > 0.f ? 1.f : (dw < 0.f ? -1.f : 0.f)) * log1pf(fabsf(dw));
  float acc[HEADSc];
#pragma unroll
  for (int h = 0; h < HEADSc; ++h) acc[h] = b2[h];
  for (int k = 0; k < 384; ++k) {
    float hv = r0 * w1[k] + r1 * w1[384 + k] + b1[k];
    hv = hv > 0.f ? hv : 0.f;
#pragma unroll
    for (int h = 0; h < HEADSc; ++h) acc[h] += hv * w2[k * HEADSc + h];
  }
#pragma unroll
  for (int h = 0; h < HEADSc; ++h) bias[(h * Lc + i) * Lc + j] = acc[h];
}

// ------------------------------------------------ QKV GEMM (gathered shifted-window rows)
// wT layout: [QKVN][DIM] f16 (pre-transposed). out: [win][qkv][head][L][HDIM] f16
__global__ void __launch_bounds__(256) k_qkv(const float* __restrict__ x,
                                             const _Float16* __restrict__ wT,
                                             const float* __restrict__ qb,
                                             _Float16* __restrict__ qkv) {
  __shared__ __align__(16) unsigned short As[32 * 32];
  __shared__ __align__(16) unsigned short Bt[64 * 32];
  __shared__ int arow[32];
  int tid = threadIdx.x, wave = tid >> 5;
  int mbase = blockIdx.x * 32, nbase = blockIdx.y * 64;
  if (tid < 32) arow[tid] = tok_srow(mbase + tid) * DIMc;
  int mi = wave >> 2, ni = wave & 3;
  int ar = tid >> 3, apart = tid & 7;   // A: 32 rows x 8 float4
  int br = tid >> 2, bpart = tid & 3;   // B: 64 rows x 4 uint4
  v8f acc = v8f_zero();
  for (int ks = 0; ks < 12; ++ks) {
    int k0 = ks * 32;
    __syncthreads();
    void* bsrc = (void*)(wT + (size_t)(nbase + br) * DIMc + k0 + bpart * 8);
#if USE_ASYNC_LDS
    __builtin_amdgcn_global_load_async_to_lds_b128(
        (gint4p)bsrc, (lint4p)&Bt[br * 32 + bpart * 8], 0, 0);
#else
    uint4 bv = *(const uint4*)bsrc;
#endif
    float4 av = *(const float4*)(x + arow[ar] + k0 + apart * 4);
    *(uint2*)&As[ar * 32 + apart * 4] = pack4h(av);
#if !USE_ASYNC_LDS
    *(uint4*)&Bt[br * 32 + bpart * 8] = bv;
#endif
    if (ks < 11) __builtin_prefetch(wT + (size_t)(nbase + br) * DIMc + k0 + 32, 0, 1);
#if USE_ASYNC_LDS
    __builtin_amdgcn_s_wait_asynccnt(0);
#endif
    __syncthreads();
    v16h a = frag_A(As + mi * 16 * 32, 32);
    v16h b = frag_A(Bt + ni * 16 * 32, 32);
    acc = wmma16(a, b, acc);
  }
  int lane = tid & 31, nn = lane & 15, hf = lane >> 4;
#pragma unroll
  for (int r = 0; r < 8; ++r) {
    int m = mbase + mi * 16 + r + hf * 8;
    int col = nbase + ni * 16 + nn;
    float vv = acc[r] + qb[col];
    int w = m / Lc, l = m % Lc;
    int which = col / DIMc, rem = col % DIMc, head = rem >> 5, d = rem & 31;
    qkv[((((size_t)w * 3 + which) * HEADSc + head) * Lc + l) * HDIMc + d] = (_Float16)vv;
  }
}

// ------------------------------------------------ attention per (window, head)
__global__ void __launch_bounds__(128) k_attn(const _Float16* __restrict__ qkv,
                                              const float* __restrict__ bias,
                                              const float* __restrict__ tau,
                                              _Float16* __restrict__ aout) {
  __shared__ __align__(16) unsigned short Qs[LPc * HDIMc];
  __shared__ __align__(16) unsigned short Ks[LPc * HDIMc];
  __shared__ __align__(16) unsigned short Vt[HDIMc * LPc];   // transposed: [d][j]
  __shared__ float Ss[LPc * LPc];
  __shared__ __align__(16) unsigned short Ps[LPc * LPc];
  __shared__ float qn[LPc], kn[LPc];
  __shared__ int rg[LPc];
  int w = blockIdx.x, head = blockIdx.y;
  int tid = threadIdx.x, wave = tid >> 5;
  size_t base = (((size_t)w * 3) * HEADSc + head) * Lc * HDIMc;
  const _Float16* qg = qkv + base;
  const _Float16* kg = qkv + base + (size_t)HEADSc * Lc * HDIMc;
  const _Float16* vg = qkv + base + 2 * (size_t)HEADSc * Lc * HDIMc;
  int wi = w % NWc, wy = wi >> 3, wx = wi & 7;
  if (tid < LPc) {  // 9-region shift-mask id per token (computed once)
    int j = tid;
    int py = wy * 7 + j / 7, px = wx * 7 + j % 7;
    rg[tid] = (j < Lc) ? ((py < 49 ? 0 : (py < 53 ? 1 : 2)) * 3 + (px < 49 ? 0 : (px < 53 ? 1 : 2))) : 0;
  }
#pragma unroll
  for (int i = 0; i < 2; ++i) {          // 256 chunks of 8 halfs per matrix
    int chunk = tid + i * 128;
    int r = chunk >> 2, part = chunk & 3;
    uint4 qv, kv, vv;
    qv.x = qv.y = qv.z = qv.w = 0u; kv = qv; vv = qv;
    if (r < Lc) {
      qv = *(const uint4*)(qg + r * HDIMc + part * 8);
      kv = *(const uint4*)(kg + r * HDIMc + part * 8);
      vv = *(const uint4*)(vg + r * HDIMc + part * 8);
    }
    *(uint4*)&Qs[chunk * 8] = qv;
    *(uint4*)&Ks[chunk * 8] = kv;
    union { uint4 q; unsigned short s[8]; } cv; cv.q = vv;
#pragma unroll
    for (int d = 0; d < 8; ++d) Vt[(part * 8 + d) * LPc + r] = cv.s[d];
  }
  __syncthreads();
  if (tid < LPc) {
    float sq = 0.f, sk = 0.f;
    for (int c = 0; c < HDIMc; ++c) {
      float a = (float)u2h(Qs[tid * HDIMc + c]); sq += a * a;
      float b = (float)u2h(Ks[tid * HDIMc + c]); sk += b * b;
    }
    qn[tid] = sqrtf(sq); kn[tid] = sqrtf(sk);
  }
  __syncthreads();
  // S = q @ k^T  (K = HDIM = 32 -> one WMMA per 16x16 tile; k rows ARE the Bt staging)
#pragma unroll
  for (int t = 0; t < 4; ++t) {
    int idx = wave * 4 + t, mi = idx >> 2, ni = idx & 3;
    v16h a = frag_A(Qs + mi * 16 * HDIMc, HDIMc);
    v16h b = frag_A(Ks + ni * 16 * HDIMc, HDIMc);
    v8f c = v8f_zero();
    c = wmma16(a, b, c);
    int lane = tid & 31, nn = lane & 15, hf = lane >> 4;
#pragma unroll
    for (int r = 0; r < 8; ++r) Ss[(mi * 16 + r + hf * 8) * LPc + ni * 16 + nn] = c[r];
  }
  __syncthreads();
  float tauh = tau[head]; if (tauh < 0.01f) tauh = 0.01f;
  float itau = 1.f / tauh;
  if (tid < LPc) {
    int i = tid;
    if (i < Lc) {
      int regi = rg[i];
      float qi = qn[i];
      const float* brow = bias + (head * Lc + i) * Lc;
      float mx = -1e30f;
      for (int j = 0; j < Lc; ++j) {
        float den = qi * kn[j]; if (den < 1e-6f) den = 1e-6f;
        float v = Ss[i * LPc + j] / den * itau + brow[j] + (regi != rg[j] ? -100.f : 0.f);
        Ss[i * LPc + j] = v;   // own row, in-place safe
        if (v > mx) mx = v;
      }
      float sum = 0.f;
      for (int j = 0; j < Lc; ++j) {
        float e = __expf(Ss[i * LPc + j] - mx);
        Ss[i * LPc + j] = e;
        sum += e;
      }
      float inv = 1.f / sum;
      for (int j = 0; j < Lc; ++j) Ps[i * LPc + j] = h2u((_Float16)(Ss[i * LPc + j] * inv));
      for (int j = Lc; j < LPc; ++j) Ps[i * LPc + j] = 0;
    } else {
      for (int j = 0; j < LPc; ++j) Ps[i * LPc + j] = 0;
    }
  }
  __syncthreads();
  // O = P @ V  (64x64 @ 64x32, 2 K-steps; Vt is the transposed B staging)
#pragma unroll
  for (int t = 0; t < 2; ++t) {
    int idx = wave * 2 + t, mi = idx >> 1, ni = idx & 1;
    v8f c = v8f_zero();
#pragma unroll
    for (int ks = 0; ks < 2; ++ks) {
      v16h a = frag_A(Ps + mi * 16 * LPc + ks * 32, LPc);
      v16h b = frag_A(Vt + (ni * 16) * LPc + ks * 32, LPc);
      c = wmma16(a, b, c);
    }
    int lane = tid & 31, nn = lane & 15, hf = lane >> 4;
#pragma unroll
    for (int r = 0; r < 8; ++r) {
      int m = mi * 16 + r + hf * 8;
      if (m < Lc)
        aout[((size_t)w * Lc + m) * DIMc + head * HDIMc + ni * 16 + nn] = (_Float16)c[r];
    }
  }
}

// ------------------------------------------------ proj GEMM + window-reverse/unshift + LN1 + residual -> x1
// wT layout: [DIM][DIM] (pre-transposed)
__global__ void __launch_bounds__(256) k_proj_ln(const _Float16* __restrict__ ain,
                                                 const _Float16* __restrict__ wT,
                                                 const float* __restrict__ pb,
                                                 const float* __restrict__ x,
                                                 const float* __restrict__ g, const float* __restrict__ bt,
                                                 float* __restrict__ x1) {
  __shared__ __align__(16) unsigned short As[16 * 32];
  __shared__ __align__(16) float ShBuf[16 * DIMc];   // aliased: Ws ushort[384*32] then Orow float[16*384]
  __shared__ float red[256];
  __shared__ float mus[16], rsg[16];
  __shared__ int srow[16];
  unsigned short* Ws = (unsigned short*)ShBuf;
  float* Orow = ShBuf;
  int tid = threadIdx.x, wave = tid >> 5;
  int mbase = blockIdx.x * 16;
  if (tid < 16) srow[tid] = tok_srow(mbase + tid) * DIMc;
  int br = tid >> 2;
  v8f acc[3]; acc[0] = v8f_zero(); acc[1] = v8f_zero(); acc[2] = v8f_zero();
  for (int ks = 0; ks < 12; ++ks) {
    int k0 = ks * 32;
    __syncthreads();
    if (tid < 64) {  // A: 16 rows x 4 uint4
      int r = tid >> 2, part = tid & 3;
      uint4 av = *(const uint4*)(ain + (size_t)(mbase + r) * DIMc + k0 + part * 8);
      *(uint4*)&As[r * 32 + part * 8] = av;
    }
#pragma unroll
    for (int i = 0; i < 6; ++i) {  // W: 384 rows x 4 uint4 = 1536 chunks
      int chunk = tid + i * 256;
      int rr = chunk >> 2, part = chunk & 3;
      uint4 wv = *(const uint4*)(wT + (size_t)rr * DIMc + k0 + part * 8);
      *(uint4*)&Ws[rr * 32 + part * 8] = wv;
    }
    if (ks < 11) __builtin_prefetch(wT + (size_t)br * DIMc + k0 + 32, 0, 1);
    __syncthreads();
    v16h a = frag_A(As, 32);
#pragma unroll
    for (int j = 0; j < 3; ++j) {
      int ncol = (wave * 3 + j) * 16;
      v16h b = frag_A(Ws + ncol * 32, 32);
      acc[j] = wmma16(a, b, acc[j]);
    }
  }
  __syncthreads();   // Ws done; Orow takes over the aliased buffer
  int lane = tid & 31, nn = lane & 15, hf = lane >> 4;
#pragma unroll
  for (int j = 0; j < 3; ++j) {
    int ncol = (wave * 3 + j) * 16 + nn;
#pragma unroll
    for (int r = 0; r < 8; ++r)
      Orow[(r + hf * 8) * DIMc + ncol] = acc[j][r] + pb[ncol];
  }
  __syncthreads();
  int row = tid >> 4, l16 = tid & 15;
  float s = 0.f;
  for (int c = l16; c < DIMc; c += 16) s += Orow[row * DIMc + c];
  red[tid] = s; __syncthreads();
  if (l16 == 0) { float t = 0.f; for (int k = 0; k < 16; ++k) t += red[row * 16 + k]; mus[row] = t * (1.f / DIMc); }
  __syncthreads();
  float mu = mus[row]; s = 0.f;
  for (int c = l16; c < DIMc; c += 16) { float d = Orow[row * DIMc + c] - mu; s += d * d; }
  red[tid] = s; __syncthreads();
  if (l16 == 0) { float t = 0.f; for (int k = 0; k < 16; ++k) t += red[row * 16 + k]; rsg[row] = rsqrtf(t * (1.f / DIMc) + 1e-5f); }
  __syncthreads();
  float rs = rsg[row];
  int so = srow[row];
  for (int c = l16; c < DIMc; c += 16)
    x1[so + c] = x[so + c] + (Orow[row * DIMc + c] - mu) * rs * g[c] + bt[c];
}

// ------------------------------------------------ fused MLP: h=gelu(x1@W1+b1); o=h@W2+b2; out = x1 + LN2(o)
// w1T: [HID][DIM], w2T: [DIM][HID] (pre-transposed)
__global__ void __launch_bounds__(256) k_mlp(const float* __restrict__ x1,
                                             const _Float16* __restrict__ w1T, const float* __restrict__ b1,
                                             const _Float16* __restrict__ w2T, const float* __restrict__ b2,
                                             const float* __restrict__ g, const float* __restrict__ bt,
                                             float* __restrict__ out) {
  __shared__ __align__(16) unsigned short Ax[16 * DIMc];     // 12KB x1 rows f16
  __shared__ __align__(16) unsigned short Hc[16 * 128];      // 4KB gelu'd hidden chunk
  __shared__ __align__(16) float ShBuf[16 * DIMc];           // aliased: Ws staging / Orow
  __shared__ float red[256];
  __shared__ float mus[16], rsg[16];
  unsigned short* Ws = (unsigned short*)ShBuf;
  float* Orow = ShBuf;
  int tid = threadIdx.x, wave = tid >> 5;
  size_t mbase = (size_t)blockIdx.x * 16;
#pragma unroll
  for (int i = 0; i < 6; ++i) {  // flat copy x1 rows -> f16 (1536 float4 chunks)
    int chunk = tid + i * 256;
    float4 fv = *(const float4*)(x1 + mbase * DIMc + chunk * 4);
    *(uint2*)&Ax[chunk * 4] = pack4h(fv);
  }
  v8f acc[3]; acc[0] = v8f_zero(); acc[1] = v8f_zero(); acc[2] = v8f_zero();
  for (int cc = 0; cc < 12; ++cc) {        // hidden chunks of 128
    v8f hacc = v8f_zero();
    for (int ks = 0; ks < 12; ++ks) {      // K over DIM
      __syncthreads();
#pragma unroll
      for (int i = 0; i < 2; ++i) {        // W1 tile: 128 rows x 4 uint4 = 512 chunks
        int chunk = tid + i * 256;
        int rr = chunk >> 2, part = chunk & 3;
        uint4 wv = *(const uint4*)(w1T + (size_t)(cc * 128 + rr) * DIMc + ks * 32 + part * 8);
        *(uint4*)&Ws[rr * 32 + part * 8] = wv;
      }
      __syncthreads();
      v16h a = frag_A(Ax + ks * 32, DIMc);
      v16h b = frag_A(Ws + (wave * 16) * 32, 32);
      hacc = wmma16(a, b, hacc);
    }
    { // bias + exact GELU, park in LDS as f16
      int lane = tid & 31, nn = lane & 15, hf = lane >> 4;
      int colg = cc * 128 + wave * 16 + nn;
#pragma unroll
      for (int r = 0; r < 8; ++r) {
        float h = hacc[r] + b1[colg];
        float ge = 0.5f * h * (1.f + erff(h * 0.70710678118f));
        Hc[(r + hf * 8) * 128 + wave * 16 + nn] = h2u((_Float16)ge);
      }
    }
    for (int ks2 = 0; ks2 < 4; ++ks2) {    // o += Hc @ W2[chunk]
      __syncthreads();
#pragma unroll
      for (int i = 0; i < 6; ++i) {        // W2 tile: 384 rows x 4 uint4 = 1536 chunks
        int chunk = tid + i * 256;
        int rr = chunk >> 2, part = chunk & 3;
        uint4 wv = *(const uint4*)(w2T + (size_t)rr * HIDc + cc * 128 + ks2 * 32 + part * 8);
        *(uint4*)&Ws[rr * 32 + part * 8] = wv;
      }
      __syncthreads();
      v16h a = frag_A(Hc + ks2 * 32, 128);
#pragma unroll
      for (int j = 0; j < 3; ++j) {
        int ncol = (wave * 3 + j) * 16;
        v16h b = frag_A(Ws + ncol * 32, 32);
        acc[j] = wmma16(a, b, acc[j]);
      }
    }
  }
  __syncthreads();   // Ws done; Orow takes over
  int lane = tid & 31, nn = lane & 15, hf = lane >> 4;
#pragma unroll
  for (int j = 0; j < 3; ++j) {
    int ncol = (wave * 3 + j) * 16 + nn;
#pragma unroll
    for (int r = 0; r < 8; ++r)
      Orow[(r + hf * 8) * DIMc + ncol] = acc[j][r] + b2[ncol];
  }
  __syncthreads();
  int row = tid >> 4, l16 = tid & 15;
  float s = 0.f;
  for (int c = l16; c < DIMc; c += 16) s += Orow[row * DIMc + c];
  red[tid] = s; __syncthreads();
  if (l16 == 0) { float t = 0.f; for (int k = 0; k < 16; ++k) t += red[row * 16 + k]; mus[row] = t * (1.f / DIMc); }
  __syncthreads();
  float mu = mus[row]; s = 0.f;
  for (int c = l16; c < DIMc; c += 16) { float d = Orow[row * DIMc + c] - mu; s += d * d; }
  red[tid] = s; __syncthreads();
  if (l16 == 0) { float t = 0.f; for (int k = 0; k < 16; ++k) t += red[row * 16 + k]; rsg[row] = rsqrtf(t * (1.f / DIMc) + 1e-5f); }
  __syncthreads();
  float rs = rsg[row];
  size_t rb = (mbase + row) * (size_t)DIMc;
  for (int c = l16; c < DIMc; c += 16)
    out[rb + c] = x1[rb + c] + (Orow[row * DIMc + c] - mu) * rs * g[c] + bt[c];
}

// ------------------------------------------------ launcher
extern "C" void kernel_launch(void* const* d_in, const int* in_sizes, int n_in,
                              void* d_out, int out_size, void* d_ws, size_t ws_size,
                              hipStream_t stream) {
  (void)in_sizes; (void)n_in; (void)out_size; (void)ws_size;
  const float* x      = (const float*)d_in[0];
  const float* qkv_w  = (const float*)d_in[1];
  const float* qkv_b  = (const float*)d_in[2];
  const float* proj_w = (const float*)d_in[3];
  const float* proj_b = (const float*)d_in[4];
  const float* tau    = (const float*)d_in[5];
  const float* mw1    = (const float*)d_in[6];
  const float* mb1    = (const float*)d_in[7];
  const float* mw2    = (const float*)d_in[8];
  const float* mb2    = (const float*)d_in[9];
  const float* n1g    = (const float*)d_in[10];
  const float* n1b    = (const float*)d_in[11];
  const float* fc1_w  = (const float*)d_in[12];
  const float* fc1_b  = (const float*)d_in[13];
  const float* fc2_w  = (const float*)d_in[14];
  const float* fc2_b  = (const float*)d_in[15];
  const float* n2g    = (const float*)d_in[16];
  const float* n2b    = (const float*)d_in[17];
  float* out = (float*)d_out;

  char* ws = (char*)d_ws;
  size_t off = 0;
  auto take = [&](size_t bytes) -> char* {
    char* p = ws + off;
    off += (bytes + 255) & ~(size_t)255;
    return p;
  };
  float*    biasT   = (float*)   take(sizeof(float) * HEADSc * Lc * Lc);
  _Float16* qkvwT   = (_Float16*)take(sizeof(_Float16) * (size_t)DIMc * QKVNc);
  _Float16* projwT  = (_Float16*)take(sizeof(_Float16) * (size_t)DIMc * DIMc);
  _Float16* fc1wT   = (_Float16*)take(sizeof(_Float16) * (size_t)DIMc * HIDc);
  _Float16* fc2wT   = (_Float16*)take(sizeof(_Float16) * (size_t)HIDc * DIMc);
  _Float16* qkvbuf  = (_Float16*)take(sizeof(_Float16) * (size_t)WINSc * 3 * HEADSc * Lc * HDIMc);
  _Float16* attnout = (_Float16*)take(sizeof(_Float16) * (size_t)TOKc * DIMc);
  float*    x1      = (float*)   take(sizeof(float) * (size_t)TOKc * DIMc);

  int n;
  n = DIMc * QKVNc; k_cvtT<<<(n + 255) / 256, 256, 0, stream>>>(qkv_w, qkvwT, DIMc, QKVNc);
  n = DIMc * DIMc;  k_cvtT<<<(n + 255) / 256, 256, 0, stream>>>(proj_w, projwT, DIMc, DIMc);
  n = DIMc * HIDc;  k_cvtT<<<(n + 255) / 256, 256, 0, stream>>>(fc1_w, fc1wT, DIMc, HIDc);
  n = HIDc * DIMc;  k_cvtT<<<(n + 255) / 256, 256, 0, stream>>>(fc2_w, fc2wT, HIDc, DIMc);
  k_meta_bias<<<(Lc * Lc + 127) / 128, 128, 0, stream>>>(mw1, mb1, mw2, mb2, biasT);
  k_qkv<<<dim3(TOKc / 32, QKVNc / 64), 256, 0, stream>>>(x, qkvwT, qkv_b, qkvbuf);
  k_attn<<<dim3(WINSc, HEADSc), 128, 0, stream>>>(qkvbuf, biasT, tau, attnout);
  k_proj_ln<<<TOKc / 16, 256, 0, stream>>>(attnout, projwT, proj_b, x, n1g, n1b, x1);
  k_mlp<<<TOKc / 16, 256, 0, stream>>>(x1, fc1wT, fc1_b, fc2wT, fc2_b, n2g, n2b, out);
}
